// ReinforcedDecoder_6184752906394
// MI455X (gfx1250) — matrix operand order repaired and checked by reading, not compiled, not run on hardware
//
#include <hip/hip_runtime.h>
#include <math.h>

// ---------------- problem constants ----------------
#define VV    50257        // vocab
#define VP    50272        // vocab padded to 16
#define DD    512
#define EE    128
#define SS    400
#define SPAD  416          // S padded to 32 for WMMA K
#define TN    60           // decode steps
#define TP    64           // T padded to 16
#define BB    16
#define MR    (TP*BB)      // 1024 padded rows, row m = t*16+b
#define KCAT  1536
#define NBLK  197          // ceil(V/256) vocab blocks for logits
#define EPS   1e-12f

typedef __bf16 bf16_t;
typedef __attribute__((ext_vector_type(16))) __bf16 v16bf;
typedef __attribute__((ext_vector_type(8)))  __bf16 v8bf;
typedef __attribute__((ext_vector_type(8)))  float  v8f;

// ---------------- WMMA fragment helpers (CDNA5 wave32 layouts) ----------------
// A 16x32 bf16: lane l (l<16): M=l, elems 0..7=K[0..7], 8..15=K[16..23]
//               lane l>=16 : M=l-16, elems 0..7=K[8..15], 8..15=K[24..31]
// B 32x16 bf16 (from row-major W[N][K]): lane l: N=l&15, elems 0..7=K[(l>>4)*8..+7],
//               8..15=K[16+(l>>4)*8..+7] (same K order as A)
// C/D 16x16 f32: VGPR r, lane l: row = (l>>4)*8 + r, col = l&15
__device__ inline v16bf mk16(v8bf lo, v8bf hi) {
  v16bf r;
#pragma unroll
  for (int i = 0; i < 8; ++i) { r[i] = lo[i]; r[i + 8] = hi[i]; }
  return r;
}
__device__ inline v16bf load_a_frag(const bf16_t* Ab, int lda, int lane) {
  int m = lane & 15, koff = (lane >> 4) * 8;
  const bf16_t* p = Ab + (size_t)m * lda + koff;
  v8bf lo = *(const v8bf*)(p);
  v8bf hi = *(const v8bf*)(p + 16);
  return mk16(lo, hi);
}
__device__ inline v16bf load_b_frag(const bf16_t* Wb, int ldb, int lane) {
  int n = lane & 15, koff = (lane >> 4) * 8;
  const bf16_t* p = Wb + (size_t)n * ldb + koff;
  v8bf lo = *(const v8bf*)(p);
  v8bf hi = *(const v8bf*)(p + 16);
  return mk16(lo, hi);
}
__device__ inline v8f wmma_bf16(v16bf a, v16bf b, v8f c) {
  return __builtin_amdgcn_wmma_f32_16x16x32_bf16(false, a, false, b,
                                                 (short)0, c, false, false);
}
__device__ inline float wred_max(float v) {
#pragma unroll
  for (int m = 16; m; m >>= 1) v = fmaxf(v, __shfl_xor(v, m));
  return v;
}
__device__ inline float wred_sum(float v) {
#pragma unroll
  for (int m = 16; m; m >>= 1) v += __shfl_xor(v, m);
  return v;
}
__device__ inline float sigm(float x) { return 1.0f / (1.0f + __expf(-x)); }

// ---------------- generic WMMA GEMM ----------------
// C[z][m][n] = op( sum_k A[z][m][k] * W[z][n][k] )    (W row-major N x K)
// grid.x*8+wave = n-tile, grid.y = m-tile, grid.z = batch. 256 thr (8 waves).
template <bool TANH, bool OUT_BF>
__global__ void k_gemm(const bf16_t* __restrict__ A, int lda, long sA,
                       const bf16_t* __restrict__ Wm, int ldb, long sW,
                       void* __restrict__ Cout, int ldc, long sC,
                       int K, int Ntiles, int Mvalid) {
  int wave = threadIdx.x >> 5, lane = threadIdx.x & 31;
  int nt = blockIdx.x * 8 + wave;
  if (nt >= Ntiles) return;
  int mt = blockIdx.y, z = blockIdx.z;
  const bf16_t* Ab = A + (size_t)z * sA + (size_t)mt * 16 * lda;
  const bf16_t* Wb = Wm + (size_t)z * sW + (size_t)nt * 16 * ldb;
  v8f acc = {0.f, 0.f, 0.f, 0.f, 0.f, 0.f, 0.f, 0.f};
  for (int k = 0; k < K; k += 32) {
    v16bf av = load_a_frag(Ab + k, lda, lane);
    v16bf bv = load_b_frag(Wb + k, ldb, lane);
    acc = wmma_bf16(av, bv, acc);
  }
  int n = lane & 15, moff = (lane >> 4) * 8;
#pragma unroll
  for (int r = 0; r < 8; ++r) {
    int m = mt * 16 + moff + r;
    float v = acc[r];
    if (TANH) v = tanhf(v);
    if (m >= Mvalid) v = 0.f;
    size_t idx = (size_t)z * sC + (size_t)m * ldc + nt * 16 + n;
    if (OUT_BF) ((bf16_t*)Cout)[idx] = (bf16_t)v;
    else        ((float*)Cout)[idx]  = v;
  }
}

// ---------------- fused logits + online-softmax partials ----------------
// Vocab block nb = 256 cols. Each wave: 4 m-tiles x 2 n-tiles (64 acc VGPRs),
// B-fragments reused 4x and A-fragments 2x from registers -> 1.5 b128 loads
// per WMMA instead of 4. launch_bounds(256,1) grants the full VGPR file;
// unroll(1) on the k-loop keeps one live fragment set; the epilogue is fully
// unrolled so acc indices stay static (no scratch spill of the accumulators).
__global__ void __launch_bounds__(256, 1)
k_logits(const bf16_t* __restrict__ cat,
         const bf16_t* __restrict__ Wout,
         const float* __restrict__ bout,
         float* __restrict__ partials) {
  __shared__ float tile[16][260];
  int wave = threadIdx.x >> 5, lane = threadIdx.x & 31;
  int nb = blockIdx.x;
  int nbase = nb * 256;
  int ncols = VV - nbase; if (ncols > 256) ncols = 256;
  // two n-tiles per wave; clamp out-of-range tiles onto row 0 (harmless: the
  // reduction below only reads columns < ncols)
  size_t r0 = (size_t)nbase + wave * 32;
  size_t r1 = r0 + 16;
  if (r0 > (size_t)(VP - 16)) r0 = 0;
  if (r1 > (size_t)(VP - 16)) r1 = 0;
  const bf16_t* Wb0 = Wout + r0 * KCAT;
  const bf16_t* Wb1 = Wout + r1 * KCAT;
  // stream-ahead hint for the weight rows this wave will consume
  __builtin_prefetch(Wb0, 0, 1);
  __builtin_prefetch(Wb1, 0, 1);
#pragma unroll 1
  for (int mg = 0; mg < TP / 4; ++mg) {
    v8f acc[4][2];
#pragma unroll
    for (int i = 0; i < 4; ++i)
#pragma unroll
      for (int j = 0; j < 2; ++j)
        acc[i][j] = (v8f){0.f, 0.f, 0.f, 0.f, 0.f, 0.f, 0.f, 0.f};
    const bf16_t* Ab = cat + (size_t)mg * 64 * KCAT;  // 4 m-tiles of 16 rows
#pragma unroll 1
    for (int k = 0; k < KCAT; k += 32) {
      v16bf b0 = load_b_frag(Wb0 + k, KCAT, lane);
      v16bf b1 = load_b_frag(Wb1 + k, KCAT, lane);
#pragma unroll
      for (int i = 0; i < 4; ++i) {
        v16bf av = load_a_frag(Ab + (size_t)i * 16 * KCAT + k, KCAT, lane);
        acc[i][0] = wmma_bf16(av, b0, acc[i][0]);
        acc[i][1] = wmma_bf16(av, b1, acc[i][1]);
      }
    }
    int n = lane & 15, moff = (lane >> 4) * 8;
    int row = threadIdx.x >> 4, sub = threadIdx.x & 15;
#pragma unroll
    for (int i = 0; i < 4; ++i) {
#pragma unroll
      for (int r = 0; r < 8; ++r) {
        tile[moff + r][wave * 32 + n]      = acc[i][0][r];
        tile[moff + r][wave * 32 + 16 + n] = acc[i][1][r];
      }
      __syncthreads();
      float mx = -3.0e38f;
      for (int c = sub; c < ncols; c += 16)
        mx = fmaxf(mx, tile[row][c] + bout[nbase + c]);
#pragma unroll
      for (int m = 8; m; m >>= 1) mx = fmaxf(mx, __shfl_xor(mx, m));
      float sm = 0.f;
      for (int c = sub; c < ncols; c += 16)
        sm += __expf(tile[row][c] + bout[nbase + c] - mx);
#pragma unroll
      for (int m = 8; m; m >>= 1) sm += __shfl_xor(sm, m);
      if (sub == 0) {
        size_t o = ((size_t)nb * MR + mg * 64 + i * 16 + row) * 2;
        partials[o] = mx; partials[o + 1] = sm;
      }
      __syncthreads();
    }
  }
}

// ---------------- prep / elementwise kernels ----------------
__global__ void k_prep_wcat(const float* Wih, const float* Whh, const float* bih,
                            const float* bhh, bf16_t* Wcat, float* bias) {
  int idx = blockIdx.x * blockDim.x + threadIdx.x;
  if (idx < 2048 * 640) {
    int n = idx / 640, k = idx % 640;
    float v = (k < EE) ? Wih[n * EE + k] : Whh[n * DD + (k - EE)];
    Wcat[idx] = (bf16_t)v;
  }
  if (idx < 2048) bias[idx] = bih[idx] + bhh[idx];
}
__global__ void k_prep_wemb(const float* We, bf16_t* out) {
  int idx = blockIdx.x * blockDim.x + threadIdx.x;
  if (idx >= VP * EE) return;
  int v = idx / EE;
  out[idx] = (v < VV) ? (bf16_t)We[idx] : (bf16_t)0.f;
}
// out[n*K+k] = in[k*N+n]  (convert + transpose to row-major N x K)
__global__ void k_prep_T(const float* in, bf16_t* out, int N, int K) {
  int idx = blockIdx.x * blockDim.x + threadIdx.x;
  if (idx >= N * K) return;
  int n = idx / K, k = idx % K;
  out[idx] = (bf16_t)in[k * N + n];
}
__global__ void k_prep_he(const float* he, bf16_t* he_bf, bf16_t* he_T) {
  int idx = blockIdx.x * blockDim.x + threadIdx.x;
  if (idx < SS * BB * DD) he_bf[idx] = (bf16_t)he[idx];
  if (idx < BB * DD * SPAD) {
    int b = idx / (DD * SPAD), r = idx % (DD * SPAD);
    int d = r / SPAD, s = r % SPAD;
    he_T[idx] = (s < SS) ? (bf16_t)he[(size_t)s * BB * DD + b * DD + d] : (bf16_t)0.f;
  }
}
__global__ void k_prep_emb(const int* tgt, const float* We, bf16_t* emb) {
  int idx = blockIdx.x * blockDim.x + threadIdx.x;
  if (idx >= TN * BB * EE) return;
  int t = idx / (BB * EE), r = idx % (BB * EE);
  int b = r / EE, e = r % EE;
  int tok = tgt[t * BB + b];
  emb[idx] = (bf16_t)We[(size_t)tok * EE + e];
}
__global__ void k_init(const float* h0, const float* c0, const bf16_t* emb,
                       bf16_t* xbuf, float* h, float* c) {
  int idx = blockIdx.x * blockDim.x + threadIdx.x;
  if (idx < BB * 640) {
    int b = idx / 640, k = idx % 640;
    xbuf[idx] = (k < EE) ? emb[b * EE + k] : (bf16_t)h0[b * DD + (k - EE)];
  }
  if (idx < BB * DD) { h[idx] = h0[idx]; c[idx] = c0[idx]; }
}
__global__ void k_cell(const float* g, const float* bias, const bf16_t* emb,
                       float* h, float* c, bf16_t* hd_bf, bf16_t* xbuf, int t) {
  int idx = blockIdx.x * blockDim.x + threadIdx.x;
  if (idx < BB * DD) {
    int b = idx / DD, d = idx % DD;
    const float* gr = g + b * 2048;
    float gi = gr[d]        + bias[d];
    float gf = gr[512 + d]  + bias[512 + d];
    float gg = gr[1024 + d] + bias[1024 + d];
    float go = gr[1536 + d] + bias[1536 + d];
    float cn = sigm(gf) * c[idx] + sigm(gi) * tanhf(gg);
    float hn = sigm(go) * tanhf(cn);
    c[idx] = cn; h[idx] = hn;
    hd_bf[(size_t)(t * BB + b) * DD + d] = (bf16_t)hn;
    xbuf[b * 640 + EE + d] = (bf16_t)hn;          // next-step input (h part)
  }
  if (t + 1 < TN && idx < BB * EE) {               // next-step input (emb part)
    int b = idx / EE, e = idx % EE;
    xbuf[b * 640 + e] = emb[(size_t)(t + 1) * BB * EE + idx];
  }
}
__global__ void k_pad_hd(bf16_t* hd_bf) {   // zero rows t=60..63
  int idx = blockIdx.x * blockDim.x + threadIdx.x;
  if (idx < (TP - TN) * BB * DD) hd_bf[(size_t)TN * BB * DD + idx] = (bf16_t)0.f;
}
__global__ void k_hdT(const bf16_t* hd_bf, bf16_t* hd_T) {
  int idx = blockIdx.x * blockDim.x + threadIdx.x;
  if (idx >= BB * DD * TP) return;
  int b = idx / (DD * TP), r = idx % (DD * TP);
  int d = r / TP, u = r % TP;
  hd_T[idx] = hd_bf[(size_t)(u * BB + b) * DD + d];
}
__global__ void k_cat_hd(const bf16_t* hd_bf, bf16_t* cat) {
  int idx = blockIdx.x * blockDim.x + threadIdx.x;
  if (idx >= MR * DD) return;
  int m = idx / DD, d = idx % DD;
  cat[(size_t)m * KCAT + d] = hd_bf[idx];
}
__global__ void k_softmax_e(const float* sE, float* Af, bf16_t* Abf) {
  int gw = (blockIdx.x * blockDim.x + threadIdx.x) >> 5;
  int lane = threadIdx.x & 31;
  if (gw >= MR) return;
  const float* row = sE + (size_t)gw * SS;
  float mx = -3.0e38f;
  for (int s = lane; s < SS; s += 32) mx = fmaxf(mx, row[s]);
  mx = wred_max(mx);
  float sm = 0.f;
  for (int s = lane; s < SS; s += 32) sm += __expf(row[s] - mx);
  sm = wred_sum(sm);
  float inv = 1.0f / sm;
  for (int s = lane; s < SPAD; s += 32) {
    if (s < SS) {
      float p = __expf(row[s] - mx) * inv;
      Af[(size_t)gw * SS + s] = p;
      Abf[(size_t)gw * SPAD + s] = (bf16_t)p;
    } else Abf[(size_t)gw * SPAD + s] = (bf16_t)0.f;
  }
}
__global__ void k_softmax_d(const float* sd, bf16_t* Abf) {
  int gw = (blockIdx.x * blockDim.x + threadIdx.x) >> 5;
  int lane = threadIdx.x & 31;
  if (gw >= MR) return;
  int t = gw / BB;
  bf16_t* out = Abf + (size_t)gw * TP;
  if (t >= TN) {                       // padded rows: all-zero attention
    for (int u = lane; u < TP; u += 32) out[u] = (bf16_t)0.f;
    return;
  }
  const float* row = sd + (size_t)gw * TP;
  float mx = -3.0e38f;
  for (int u = lane; u <= t; u += 32) mx = fmaxf(mx, row[u]);
  mx = wred_max(mx);
  float sm = 0.f;
  for (int u = lane; u <= t; u += 32) sm += __expf(row[u] - mx);
  sm = wred_sum(sm);
  float inv = 1.0f / sm;
  for (int u = lane; u < TP; u += 32)
    out[u] = (u <= t) ? (bf16_t)(__expf(row[u] - mx) * inv) : (bf16_t)0.f;
}
__global__ void k_fix_cd0(bf16_t* cat) {  // c_d row t=0 := 0
  int idx = blockIdx.x * blockDim.x + threadIdx.x;
  if (idx >= BB * DD) return;
  int b = idx / DD, d = idx % DD;
  cat[(size_t)b * KCAT + 1024 + d] = (bf16_t)0.f;
}
__global__ void k_pswitch(const bf16_t* cat, const float* Wu, const float* bu,
                          float* ps) {
  int gw = (blockIdx.x * blockDim.x + threadIdx.x) >> 5;
  int lane = threadIdx.x & 31;
  if (gw >= TN * BB) return;
  const bf16_t* row = cat + (size_t)gw * KCAT;
  float acc = 0.f;
  for (int j = lane; j < KCAT; j += 32) acc += (float)row[j] * Wu[j];
  acc = wred_sum(acc);
  if (lane == 0) ps[gw] = sigm(acc + bu[0]);
}
__global__ void k_tgt_logit(const bf16_t* cat, const bf16_t* Wout,
                            const float* bout, const int* tgt, float* zt) {
  int gw = (blockIdx.x * blockDim.x + threadIdx.x) >> 5;
  int lane = threadIdx.x & 31;
  if (gw >= TN * BB) return;
  int v = tgt[gw];  // row m=t*16+b matches tgt[t][b] layout
  const bf16_t* a = cat + (size_t)gw * KCAT;
  const bf16_t* w = Wout + (size_t)v * KCAT;
  float acc = 0.f;
  for (int j = lane; j < KCAT; j += 32) acc += (float)a[j] * (float)w[j];
  acc = wred_sum(acc);
  if (lane == 0) zt[gw] = acc + bout[v];
}
__global__ void k_logZ(const float* partials, float* logZ) {
  int gw = (blockIdx.x * blockDim.x + threadIdx.x) >> 5;
  int lane = threadIdx.x & 31;
  if (gw >= TN * BB) return;
  float mx = -3.0e38f;
  for (int nb = lane; nb < NBLK; nb += 32)
    mx = fmaxf(mx, partials[((size_t)nb * MR + gw) * 2]);
  mx = wred_max(mx);
  float s = 0.f;
  for (int nb = lane; nb < NBLK; nb += 32) {
    size_t o = ((size_t)nb * MR + gw) * 2;
    s += partials[o + 1] * __expf(partials[o] - mx);
  }
  s = wred_sum(s);
  if (lane == 0) logZ[gw] = mx + __logf(s);
}
__global__ void k_zero_out(float* out) { if (threadIdx.x == 0) out[0] = 0.f; }
__global__ void k_loss(const float* Af, const float* align, const int* tgt,
                       const float* ps, const float* zt, const float* logZ,
                       float* out) {
  int gw = (blockIdx.x * blockDim.x + threadIdx.x) >> 5;
  int lane = threadIdx.x & 31;
  if (gw >= TN * BB) return;
  float cs = 0.f;
  for (int s = lane; s < SS; s += 32)
    cs += Af[(size_t)gw * SS + s] * align[(size_t)gw * SS + s];
  cs = wred_sum(cs);
  if (lane == 0) {
    int targ = tgt[gw];
    float p = ps[gw];
    float tg = (1.0f - p) * __expf(zt[gw] - logZ[gw]);
    float tc = p * cs + EPS;
    float l = (targ != 0) ? -logf(tg + tc + EPS) : 0.f;
    atomicAdd(out, l);
  }
}

// ---------------- host launch ----------------
static inline size_t alup(size_t x) { return (x + 255) & ~(size_t)255; }

extern "C" void kernel_launch(void* const* d_in, const int* in_sizes, int n_in,
                              void* d_out, int out_size, void* d_ws, size_t ws_size,
                              hipStream_t stream) {
  (void)in_sizes; (void)n_in; (void)out_size; (void)ws_size;
  const int*   tgt  = (const int*)d_in[0];
  // d_in[1] = src (unused by the loss)
  const float* align= (const float*)d_in[2];
  const float* h_e  = (const float*)d_in[3];
  const float* h0   = (const float*)d_in[4];
  const float* c0   = (const float*)d_in[5];
  const float* Wemb = (const float*)d_in[6];
  const float* Wih  = (const float*)d_in[7];
  const float* Whh  = (const float*)d_in[8];
  const float* bih  = (const float*)d_in[9];
  const float* bhh  = (const float*)d_in[10];
  const float* Wae  = (const float*)d_in[11];
  const float* Wad  = (const float*)d_in[12];
  const float* Wproj= (const float*)d_in[13];
  const float* Wu   = (const float*)d_in[14];
  const float* bu   = (const float*)d_in[15];
  const float* bout = (const float*)d_in[16];
  float* out = (float*)d_out;

  // workspace carve-out (~190 MB; W_out_bf dominates and stays L2-resident)
  char* p = (char*)d_ws; size_t off = 0;
  auto take = [&](size_t bytes) { char* r = p + off; off = alup(off + bytes); return (void*)r; };
  bf16_t* Wout_bf  = (bf16_t*)take((size_t)VP * KCAT * 2);
  bf16_t* Wemb_bf  = (bf16_t*)take((size_t)VP * EE * 2);
  bf16_t* WprojT   = (bf16_t*)take((size_t)KCAT * EE * 2);
  bf16_t* Wcat     = (bf16_t*)take((size_t)2048 * 640 * 2);
  float*  bias     = (float*) take(2048 * 4);
  bf16_t* WaeT     = (bf16_t*)take((size_t)DD * DD * 2);
  bf16_t* WadT     = (bf16_t*)take((size_t)DD * DD * 2);
  bf16_t* he_bf    = (bf16_t*)take((size_t)SS * BB * DD * 2);
  bf16_t* heT      = (bf16_t*)take((size_t)BB * DD * SPAD * 2);
  bf16_t* emb_bf   = (bf16_t*)take((size_t)TN * BB * EE * 2);
  bf16_t* xbuf     = (bf16_t*)take((size_t)BB * 640 * 2);
  float*  gbuf     = (float*) take((size_t)BB * 2048 * 4);
  float*  hstate   = (float*) take((size_t)BB * DD * 4);
  float*  cstate   = (float*) take((size_t)BB * DD * 4);
  bf16_t* hd_bf    = (bf16_t*)take((size_t)MR * DD * 2);
  bf16_t* hdT      = (bf16_t*)take((size_t)BB * DD * TP * 2);
  bf16_t* qe       = (bf16_t*)take((size_t)MR * DD * 2);
  bf16_t* qd       = (bf16_t*)take((size_t)MR * DD * 2);
  float*  sE       = (float*) take((size_t)MR * SS * 4);
  float*  Ae_f     = (float*) take((size_t)MR * SS * 4);
  bf16_t* Ae_bf    = (bf16_t*)take((size_t)MR * SPAD * 2);
  float*  sdbuf    = (float*) take((size_t)MR * TP * 4);
  bf16_t* Ad_bf    = (bf16_t*)take((size_t)MR * TP * 2);
  bf16_t* cat      = (bf16_t*)take((size_t)MR * KCAT * 2);
  float*  ps       = (float*) take((size_t)MR * 4);
  float*  zt       = (float*) take((size_t)MR * 4);
  float*  logZ     = (float*) take((size_t)MR * 4);
  float*  partials = (float*) take((size_t)NBLK * MR * 2 * 4);

  auto grid1 = [](int total) { return dim3((total + 255) / 256); };

  // --- prep: bf16 conversions / transposes ---
  k_prep_wcat<<<grid1(2048 * 640), 256, 0, stream>>>(Wih, Whh, bih, bhh, Wcat, bias);
  k_prep_wemb<<<grid1(VP * EE), 256, 0, stream>>>(Wemb, Wemb_bf);
  k_prep_T<<<grid1(KCAT * EE), 256, 0, stream>>>(Wproj, WprojT, KCAT, EE);
  k_prep_T<<<grid1(DD * DD), 256, 0, stream>>>(Wae, WaeT, DD, DD);
  k_prep_T<<<grid1(DD * DD), 256, 0, stream>>>(Wad, WadT, DD, DD);
  k_prep_he<<<grid1(BB * DD * SPAD), 256, 0, stream>>>(h_e, he_bf, heT);
  k_prep_emb<<<grid1(TN * BB * EE), 256, 0, stream>>>(tgt, Wemb, emb_bf);
  k_init<<<grid1(BB * 640), 256, 0, stream>>>(h0, c0, emb_bf, xbuf, hstate, cstate);

  // --- W_out = tanh(W_emb @ W_proj), bf16 out, L2-resident ---
  k_gemm<true, true><<<dim3(12, VP / 16, 1), 256, 0, stream>>>(
      Wemb_bf, EE, 0, WprojT, EE, 0, Wout_bf, KCAT, 0, EE, KCAT / 16, VV);

  // --- LSTM: 60 sequential (gates GEMM + cell) pairs ---
  for (int t = 0; t < TN; ++t) {
    k_gemm<false, false><<<dim3(16, 1, 1), 256, 0, stream>>>(
        xbuf, 640, 0, Wcat, 640, 0, gbuf, 2048, 0, 640, 128, BB);
    k_cell<<<grid1(BB * DD), 256, 0, stream>>>(gbuf, bias, emb_bf, hstate,
                                               cstate, hd_bf, xbuf, t);
  }
  k_pad_hd<<<grid1((TP - TN) * BB * DD), 256, 0, stream>>>(hd_bf);
  k_hdT<<<grid1(BB * DD * TP), 256, 0, stream>>>(hd_bf, hdT);
  k_cat_hd<<<grid1(MR * DD), 256, 0, stream>>>(hd_bf, cat);

  // --- q_e = hd@Wae, q_d = hd@Wad ---
  k_gemm<false, true><<<dim3(4, TP * BB / 16, 1), 256, 0, stream>>>(
      hd_bf, DD, 0, WaeT, DD, 0, qe, DD, 0, DD, DD / 16, MR);
  k_gemm<false, true><<<dim3(4, TP * BB / 16, 1), 256, 0, stream>>>(
      hd_bf, DD, 0, WadT, DD, 0, qd, DD, 0, DD, DD / 16, MR);

  // --- encoder attention: scores (batched over B), softmax, c_e -> cat[:,512:1024] ---
  k_gemm<false, false><<<dim3(4, TP / 16, BB), 256, 0, stream>>>(
      qe, BB * DD, DD, he_bf, BB * DD, DD, sE, BB * SS, SS, DD, SS / 16, TP);
  k_softmax_e<<<grid1(MR * 32), 256, 0, stream>>>(sE, Ae_f, Ae_bf);
  k_gemm<false, true><<<dim3(4, TP / 16, BB), 256, 0, stream>>>(
      Ae_bf, BB * SPAD, SPAD, heT, SPAD, (long)DD * SPAD,
      cat + 512, BB * KCAT, KCAT, SPAD, DD / 16, TP);

  // --- decoder self-attention: sd, causal softmax, c_d -> cat[:,1024:1536] ---
  k_gemm<false, false><<<dim3(1, TP / 16, BB), 256, 0, stream>>>(
      qd, BB * DD, DD, hd_bf, BB * DD, DD, sdbuf, BB * TP, TP, DD, TP / 16, TP);
  k_softmax_d<<<grid1(MR * 32), 256, 0, stream>>>(sdbuf, Ad_bf);
  k_gemm<false, true><<<dim3(4, TP / 16, BB), 256, 0, stream>>>(
      Ad_bf, BB * TP, TP, hdT, TP, (long)DD * TP,
      cat + 1024, BB * KCAT, KCAT, TP, DD / 16, TP);
  k_fix_cd0<<<grid1(BB * DD), 256, 0, stream>>>(cat);

  // --- switch prob, fused logits softmax partials, target logit, logZ, loss ---
  k_pswitch<<<grid1(TN * BB * 32), 256, 0, stream>>>(cat, Wu, bu, ps);
  k_logits<<<dim3(NBLK), 256, 0, stream>>>(cat, Wout_bf, bout, partials);
  k_tgt_logit<<<grid1(TN * BB * 32), 256, 0, stream>>>(cat, Wout_bf, bout, tgt, zt);
  k_logZ<<<grid1(TN * BB * 32), 256, 0, stream>>>(partials, logZ);
  k_zero_out<<<1, 64, 0, stream>>>(out);
  k_loss<<<grid1(TN * BB * 32), 256, 0, stream>>>(Ae_f, align, tgt, ps, zt, logZ, out);
}
